// ARConv_317827580492
// MI455X (gfx1250) — matrix operand order repaired and checked by reading, not compiled, not run on hardware
//
#include <hip/hip_runtime.h>
#include <math.h>

// ---------------------------------------------------------------------------
// ARConv forward on gfx1250 (MI455X, wave32, WMMA).
// Implicit GEMMs (M=102400 px, N=64, K=576) on v_wmma_f32_16x16x32_f16.
// K-dimension is permuted (pi) so all per-lane A addresses are compile-time
// constants. f16 intermediates let conv staging be a raw halo copy, done via
// the CDNA5 Tensor Data Mover (tensor_load_to_lds, TENSORcnt) or the async
// global->LDS path (global_load_async_to_lds_b32, ASYNCcnt).
// ---------------------------------------------------------------------------

typedef _Float16 v16h __attribute__((ext_vector_type(16)));
typedef float    v8f  __attribute__((ext_vector_type(8)));
typedef unsigned int u32x4 __attribute__((ext_vector_type(4)));
typedef int          i32x8 __attribute__((ext_vector_type(8)));
typedef int          i32x4 __attribute__((ext_vector_type(4)));

#define BN_   4
#define CN_   64
#define HN_   160
#define WN_   160
#define HW_   (HN_ * WN_)            // 25600
#define PIX_  (BN_ * HW_)            // 102400
#define HP_   (HN_ + 2)              // 162
#define PLP_  (HP_ * HP_)            // 26244
#define XPAD_ELEMS (BN_ * CN_ * PLP_)
#define KSTEPS_ 18                   // 576/32
#define MTILES_ (PIX_ / 16)          // 6400
#define EPSBN_ 1e-5f

// LDS tile: 192 rows (ci*3+rr) x 164 halfs (row stride 164 keeps DMA chunks
// 4B-aligned; valid cc in [0,161] stored at +1 => data at +2..+161+... ).
#define ROWS_    (CN_ * 3)           // 192
#define RSTRIDE_ 164
#define TIN_ELEMS (ROWS_ * RSTRIDE_) // 31488 halfs = 62976 B
#define HLOFF_   (32 * 3 * RSTRIDE_) // +32 channels for lane-half 1

#define STAGE_CONVERT 0
#define STAGE_ASYNC   1
#define STAGE_TDM     2

__device__ __forceinline__ float act_apply(float v, int act) {
    if (act == 1) return v >= 0.f ? v : 0.01f * v;   // LeakyReLU(0.01)
    if (act == 2) return tanhf(v);
    return v;
}

__device__ __forceinline__ v8f vzero8() {
    v8f z;
#pragma unroll
    for (int i = 0; i < 8; ++i) z[i] = 0.f;
    return z;
}

// ---------------------------------------------------------------------------
// TDM: DMA one 160-half image line for all 64 channels into LDS.
// D# (ISA 8.3-8.6): count=1, type=2(image), data_size=1(2B), iterate_enable,
// tile 160x1, tensor_dim0=stride0=160, iterate_count=63 (64 iters),
// global_addr += HW per iter (channel stride), lds_addr += 3*164 halfs.
// This toolchain exposes the 6-arg builtin: (g0, g1, g2, g3, g4, cpol).
// ---------------------------------------------------------------------------
__device__ __forceinline__ void tdm_load_line(const _Float16* gsrc, unsigned lds_byte) {
    unsigned long long ga = (unsigned long long)(uintptr_t)gsrc;
    u32x4 g0 = {
        1u,                                            // count=1, user D#
        lds_byte,                                      // lds_addr (bytes)
        (unsigned)(ga & 0xFFFFFFFFu),                  // global_addr[31:0]
        (unsigned)((ga >> 32) & 0x1FFFFFFu) | 0x80000000u  // [56:32] | type=2
    };
    i32x8 g1 = {
        (1 << 16) | (1 << 19),       // data_size=2B | iterate_enable
        (int)(160u << 16),           // tensor_dim0[15:0] @bit48
        (int)(1u << 16),             // tensor_dim0[31:16]=0 | tensor_dim1 lo=1
        (int)(160u << 16),           // tensor_dim1 hi=0 | tile_dim0=160
        1,                           // tile_dim1=1 | tile_dim2=0
        160,                         // tensor_dim0_stride[31:0]
        (int)(160u << 16),           // stride0[47:32]=0 | stride1[15:0]=160
        0                            // stride1[47:16]=0
    };
    i32x4 g2 = {
        64,                          // tensor_dim2 (unused with iterate)
        3 * RSTRIDE_,                // lds_addr_increment (data_size units)
        HW_,                         // global_addr_increment lo32 (units)
        (int)(63u << 16)             // gaddr_inc hi=0 | iterate_count=63
    };
    i32x4 g3 = {0, 0, 0, 0};
    i32x8 g4 = {0, 0, 0, 0, 0, 0, 0, 0};
    __builtin_amdgcn_tensor_load_to_lds(g0, g1, g2, g3, g4, 0);
}

// ---------------------------------------------------------------------------
// Pad x (B,64,160,160) -> x_pad (B,64,162,162) with a 1-pixel zero border.
// ---------------------------------------------------------------------------
__global__ void pad_kernel(const float* __restrict__ x, float* __restrict__ xp) {
    int idx = blockIdx.x * blockDim.x + threadIdx.x;
    if (idx >= XPAD_ELEMS) return;
    int wp = idx % HP_;
    int t  = idx / HP_;
    int hp = t % HP_;
    int bc = t / HP_;
    float v = 0.f;
    if (hp >= 1 && hp < HP_ - 1 && wp >= 1 && wp < HP_ - 1)
        v = x[(bc * HN_ + (hp - 1)) * WN_ + (wp - 1)];
    xp[idx] = v;
}

// ---------------------------------------------------------------------------
// Pack fp32 OIHW (64,64,3,3) weights into B fragments under permutation pi:
// u = ((kappa>>5)<<4)|(kappa&15); ci = u/9 + 32*bit4(kappa); tap = u%9.
// B-fragment lane mapping: kappa = s*32 + j + 16*(lane>>4).
// dst layout: [ntile(4)][step(18)][lane(32)][j(16)] halfs.
// ---------------------------------------------------------------------------
__global__ void pack_w_kernel(const float* __restrict__ w, _Float16* __restrict__ dst) {
    int t = blockIdx.x * blockDim.x + threadIdx.x;
    if (t >= 4 * KSTEPS_ * 32 * 16) return;
    int j    = t & 15;
    int lane = (t >> 4) & 31;
    int rest = t >> 9;
    int s    = rest % KSTEPS_;
    int nt   = rest / KSTEPS_;
    int kap  = s * 32 + j + 16 * (lane >> 4);
    int u    = ((kap >> 5) << 4) | (kap & 15);
    int ci   = u / 9 + 32 * ((kap >> 4) & 1);
    int tap  = u % 9;
    int kh   = tap / 3;
    int kw   = tap - kh * 3;
    int co   = nt * 16 + (lane & 15);
    dst[t] = (_Float16)w[((co * CN_ + ci) * 3 + kh) * 3 + kw];
}

// ---------------------------------------------------------------------------
// Fold conv bias (+ optional eval-mode BN) into per-channel scale/shift.
// ---------------------------------------------------------------------------
__global__ void prep_affine_kernel(const float* __restrict__ bias,
                                   const float* __restrict__ g,
                                   const float* __restrict__ be,
                                   const float* __restrict__ mu,
                                   const float* __restrict__ va,
                                   float* __restrict__ sc, float* __restrict__ sh) {
    int c = threadIdx.x;
    if (c >= CN_) return;
    if (g) {
        float s = g[c] / sqrtf(va[c] + EPSBN_);
        sc[c] = s;
        sh[c] = (bias[c] - mu[c]) * s + be[c];
    } else {
        sc[c] = 1.f;
        sh[c] = bias[c];
    }
}

// ---------------------------------------------------------------------------
// 3x3 pad-1 conv C64->C64, one block (320 thr = 10 waves) per output row.
// STAGE_CONVERT: fp32 input, bounds + prescale + f32->f16 staging.
// STAGE_ASYNC  : f16 input, per-lane global_load_async_to_lds_b32.
// STAGE_TDM    : f16 input, 3 tensor_load_to_lds DMAs (wave 0 issues).
// Compute: fully unrolled; every A element is ds_load_u16 @ immediate offset.
// ---------------------------------------------------------------------------
template <typename TIN, typename TOUT, int STAGE>
__global__ __launch_bounds__(320)
void conv3x3_wmma_kernel(const TIN* __restrict__ in,
                         const _Float16* __restrict__ wp,
                         const float* __restrict__ sc,
                         const float* __restrict__ sh,
                         TOUT* __restrict__ out,
                         float prescale, int act) {
    __shared__ _Float16 tin[TIN_ELEMS];

    const int tid = threadIdx.x;
    const int b   = blockIdx.x / HN_;
    const int h   = blockIdx.x - b * HN_;

    if constexpr (STAGE == STAGE_TDM) {
        // Border columns (cc=0,161) for every row.
        for (int i = tid; i < ROWS_; i += 320) {
            tin[i * RSTRIDE_ + 1]   = (_Float16)0.f;
            tin[i * RSTRIDE_ + 162] = (_Float16)0.f;
        }
        // Out-of-image rows zeroed (h==0 / h==HN-1 blocks only).
#pragma unroll
        for (int rr = 0; rr < 3; ++rr) {
            int hh = h + rr - 1;
            if (hh < 0 || hh >= HN_) {
                for (int i = tid; i < CN_ * 160; i += 320) {
                    int ci = i / 160, k = i - ci * 160;
                    tin[(ci * 3 + rr) * RSTRIDE_ + 2 + k] = (_Float16)0.f;
                }
            }
        }
        // TDM DMAs: one per valid halo row, issued by wave 0 only.
        if (tid < 32) {
#pragma unroll
            for (int rr = 0; rr < 3; ++rr) {
                int hh = h + rr - 1;
                if (hh >= 0 && hh < HN_) {
                    const TIN* gp = in + ((size_t)b * CN_ * HN_ + hh) * WN_;
                    unsigned lb = (unsigned)(uintptr_t)&tin[rr * RSTRIDE_ + 2];
                    tdm_load_line(gp, lb);
                }
            }
            __builtin_amdgcn_s_wait_tensorcnt(0);
        }
    } else if constexpr (STAGE == STAGE_ASYNC) {
        // 192 rows x (80 b32 chunks + 1 border task).
        for (int idx = tid; idx < ROWS_ * 81; idx += 320) {
            int row = idx / 81;
            int c   = idx - row * 81;
            int rr  = row % 3;
            int ci  = row / 3;
            int hh  = h + rr - 1;
            int lb  = row * RSTRIDE_;
            if (c == 80) {
                tin[lb + 1]   = (_Float16)0.f;     // cc = 0
                tin[lb + 162] = (_Float16)0.f;     // cc = 161
            } else if (hh >= 0 && hh < HN_) {
                const TIN* gp = in + ((size_t)(b * CN_ + ci) * HN_ + hh) * WN_ + 2 * c;
                unsigned lo = (unsigned)(uintptr_t)&tin[lb + 2 + 2 * c];
                asm volatile("global_load_async_to_lds_b32 %0, %1, off"
                             :: "v"(lo), "v"(gp) : "memory");
            } else {
                tin[lb + 2 + 2 * c] = (_Float16)0.f;
                tin[lb + 3 + 2 * c] = (_Float16)0.f;
            }
        }
        asm volatile("s_wait_asynccnt 0" ::: "memory");
    } else {
        // Convert staging (fp32 input, fused prescale).
        for (int i = tid; i < ROWS_ * HP_; i += 320) {
            int cc  = i % HP_;
            int row = i / HP_;
            int rr  = row % 3;
            int ci  = row / 3;
            int hh  = h + rr - 1;
            int ww  = cc - 1;
            float v = 0.f;
            if (hh >= 0 && hh < HN_ && ww >= 0 && ww < WN_)
                v = (float)in[((b * CN_ + ci) * HN_ + hh) * WN_ + ww] * prescale;
            tin[row * RSTRIDE_ + cc + 1] = (_Float16)v;
        }
    }
    __syncthreads();

    const int wave = tid >> 5;           // 0..9 -> 16-px tile in this row
    const int lane = tid & 31;
    const int hl   = lane >> 4;
    const int r    = lane & 15;
    // Lane base: pixel column + lane-half channel offset. All further A
    // offsets are compile-time constants (pi-permuted K).
    const _Float16* tb = tin + hl * HLOFF_ + wave * 16 + r + 1;

    v8f acc[4];
#pragma unroll
    for (int nt = 0; nt < 4; ++nt) acc[nt] = vzero8();

#pragma unroll
    for (int s = 0; s < KSTEPS_; ++s) {
        v16h a;
#pragma unroll
        for (int j = 0; j < 16; ++j) {
            const int u   = s * 16 + j;          // constant after unroll
            const int cil = u / 9;
            const int tap = u - cil * 9;
            const int kh  = tap / 3;
            const int kw  = tap - kh * 3;
            a[j] = tb[(cil * 3 + kh) * RSTRIDE_ + kw];
        }
#pragma unroll
        for (int nt = 0; nt < 4; ++nt) {
            const v16h bf = *reinterpret_cast<const v16h*>(
                wp + ((nt * KSTEPS_ + s) * 32 + lane) * 16);
            acc[nt] = __builtin_amdgcn_wmma_f32_16x16x32_f16(
                false, a, false, bf, (short)0, acc[nt], false, false);
        }
    }

    // Epilogue: lane holds column N=r (out channel), rows M=i+8*hl (pixels).
#pragma unroll
    for (int nt = 0; nt < 4; ++nt) {
        int co = nt * 16 + r;
        float scv = sc[co], shv = sh[co];
#pragma unroll
        for (int i = 0; i < 8; ++i) {
            int wo = wave * 16 + i + 8 * hl;
            float v = act_apply(acc[nt][i] * scv + shv, act);
            out[((b * CN_ + co) * HN_ + h) * WN_ + wo] = (TOUT)v;
        }
    }
}

// ---------------------------------------------------------------------------
// l/w heads on f16 feature map: conv3x3(C64->1)+BN+LReLU -> conv1x1+BN ->
// sigmoid -> *17+1. Tiny FLOPs: plain VALU kernel, both heads fused.
// ---------------------------------------------------------------------------
__global__ __launch_bounds__(256)
void heads_kernel(const _Float16* __restrict__ f,
                  const float* lw1, const float* lb1, const float* lg1, const float* lbe1,
                  const float* lmu1, const float* lva1,
                  const float* lw2, const float* lb2, const float* lg2, const float* lbe2,
                  const float* lmu2, const float* lva2,
                  const float* ww1, const float* wb1, const float* wg1, const float* wbe1,
                  const float* wmu1, const float* wva1,
                  const float* ww2, const float* wb2, const float* wg2, const float* wbe2,
                  const float* wmu2, const float* wva2,
                  float* __restrict__ lmap, float* __restrict__ wmap) {
    int p = blockIdx.x * blockDim.x + threadIdx.x;
    if (p >= PIX_) return;
    int b   = p / HW_;
    int rem = p - b * HW_;
    int h   = rem / WN_;
    int w   = rem - h * WN_;

    float suml = 0.f, sumw = 0.f;
    for (int ci = 0; ci < CN_; ++ci) {
        const _Float16* fp = f + (size_t)(b * CN_ + ci) * HW_;
#pragma unroll
        for (int kh = 0; kh < 3; ++kh) {
#pragma unroll
            for (int kw = 0; kw < 3; ++kw) {
                int hh = h + kh - 1, ww = w + kw - 1;
                if (hh < 0 || hh >= HN_ || ww < 0 || ww >= WN_) continue;
                float v = (float)fp[hh * WN_ + ww];
                suml += v * lw1[(ci * 3 + kh) * 3 + kw];
                sumw += v * ww1[(ci * 3 + kh) * 3 + kw];
            }
        }
    }
    {
        float v = suml + lb1[0];
        float s1 = lg1[0] / sqrtf(lva1[0] + EPSBN_);
        v = (v - lmu1[0]) * s1 + lbe1[0];
        v = v >= 0.f ? v : 0.01f * v;
        v = v * lw2[0] + lb2[0];
        float s2 = lg2[0] / sqrtf(lva2[0] + EPSBN_);
        v = (v - lmu2[0]) * s2 + lbe2[0];
        v = 1.f / (1.f + expf(-v));
        lmap[p] = v * 17.f + 1.f;
    }
    {
        float v = sumw + wb1[0];
        float s1 = wg1[0] / sqrtf(wva1[0] + EPSBN_);
        v = (v - wmu1[0]) * s1 + wbe1[0];
        v = v >= 0.f ? v : 0.01f * v;
        v = v * ww2[0] + wb2[0];
        float s2 = wg2[0] / sqrtf(wva2[0] + EPSBN_);
        v = (v - wmu2[0]) * s2 + wbe2[0];
        v = 1.f / (1.f + expf(-v));
        wmap[p] = v * 17.f + 1.f;
    }
}

// ---------------------------------------------------------------------------
// Fused bilinear sampling + stride-3 main conv + (out*m + bias).
// out[co,p] = sum_{ci,n} W[co,ci,n] * samp[ci,n,p]; the sample index n is a
// compile-time constant under pi, so the 9 per-lane coefficient records stay
// in registers; hl only offsets the x_pad channel base.
// ---------------------------------------------------------------------------
struct Coef { int i0, i1, i2, i3; float g0, g1, g2, g3; };

__global__ __launch_bounds__(128)
void sample_mainconv_kernel(const float* __restrict__ xpad,
                            const float* __restrict__ lmap,
                            const float* __restrict__ wmap,
                            const _Float16* __restrict__ wp,
                            const float* __restrict__ mbuf,
                            const float* __restrict__ bbuf,
                            float* __restrict__ out) {
    __shared__ Coef coefs[4][144];   // 4 waves * 16 px * 9 samples

    const int wave  = threadIdx.x >> 5;
    const int lane  = threadIdx.x & 31;
    const int hl    = lane >> 4;
    const int r     = lane & 15;
    const int mtile = blockIdx.x * 4 + wave;

    int b = 0, h = 0, wb = 0;
    if (mtile < MTILES_) {
        const int pix0 = mtile * 16;
        b  = pix0 / HW_;
        int rem = pix0 - b * HW_;
        h  = rem / WN_;
        wb = rem - h * WN_;
        for (int e = lane; e < 144; e += 32) {
            int rr = e / 9;
            int n  = e - rr * 9;
            int wc = wb + rr;
            int p  = (b * HN_ + h) * WN_ + wc;
            float lv = lmap[p];
            float wv = wmap[p];
            float pnx = (float)(n / 3) - 1.f;
            float pny = (float)(n % 3) - 1.f;
            float px = (float)(h + 1)  + (lv * (1.f / 3.f)) * pnx;
            float py = (float)(wc + 1) + (wv * (1.f / 3.f)) * pny;
            float fx = floorf(px), fy = floorf(py);
            float qltx = fminf(fmaxf(fx, 0.f), (float)(HP_ - 1));
            float qlty = fminf(fmaxf(fy, 0.f), (float)(HP_ - 1));
            float qrbx = fminf(fmaxf(fx + 1.f, 0.f), (float)(HP_ - 1));
            float qrby = fminf(fmaxf(fy + 1.f, 0.f), (float)(HP_ - 1));
            float pxc = fminf(fmaxf(px, 0.f), (float)(HP_ - 1));
            float pyc = fminf(fmaxf(py, 0.f), (float)(HP_ - 1));
            Coef c;
            c.g0 = (1.f + (qltx - pxc)) * (1.f + (qlty - pyc));   // lt
            c.g1 = (1.f - (qrbx - pxc)) * (1.f - (qrby - pyc));   // rb
            c.g2 = (1.f + (qltx - pxc)) * (1.f - (qrby - pyc));   // lb
            c.g3 = (1.f - (qrbx - pxc)) * (1.f + (qlty - pyc));   // rt
            c.i0 = (int)qltx * HP_ + (int)qlty;
            c.i1 = (int)qrbx * HP_ + (int)qrby;
            c.i2 = (int)qltx * HP_ + (int)qrby;
            c.i3 = (int)qrbx * HP_ + (int)qlty;
            coefs[wave][e] = c;
        }
    }
    __syncthreads();
    if (mtile >= MTILES_) return;

    Coef cf[9];
#pragma unroll
    for (int n = 0; n < 9; ++n) cf[n] = coefs[wave][r * 9 + n];

    v8f acc[4];
#pragma unroll
    for (int nt = 0; nt < 4; ++nt) acc[nt] = vzero8();

    const float* xbh = xpad + (size_t)b * CN_ * PLP_ + (size_t)hl * 32 * PLP_;

#pragma unroll
    for (int s = 0; s < KSTEPS_; ++s) {
        v16h a;
#pragma unroll
        for (int j = 0; j < 16; ++j) {
            const int u   = s * 16 + j;          // constant after unroll
            const int cil = u / 9;
            const int n   = u - cil * 9;
            const Coef& c = cf[n];
            const float* xp = xbh + (size_t)cil * PLP_;
            float v = c.g0 * xp[c.i0] + c.g1 * xp[c.i1] +
                      c.g2 * xp[c.i2] + c.g3 * xp[c.i3];
            a[j] = (_Float16)v;
        }
#pragma unroll
        for (int nt = 0; nt < 4; ++nt) {
            const v16h bf = *reinterpret_cast<const v16h*>(
                wp + ((nt * KSTEPS_ + s) * 32 + lane) * 16);
            acc[nt] = __builtin_amdgcn_wmma_f32_16x16x32_f16(
                false, a, false, bf, (short)0, acc[nt], false, false);
        }
    }

#pragma unroll
    for (int nt = 0; nt < 4; ++nt) {
        int co = nt * 16 + r;
#pragma unroll
        for (int i = 0; i < 8; ++i) {
            int wo = wb + i + 8 * hl;
            size_t oidx = ((size_t)(b * CN_ + co) * HN_ + h) * WN_ + wo;
            out[oidx] = acc[nt][i] * mbuf[oidx] + bbuf[oidx];
        }
    }
}

// ---------------------------------------------------------------------------
// Host-side orchestration.
// d_in order = recursive insertion-order flatten of setup_inputs():
//  0:x | 1-6:m (w,b)x3 | 7-12:b (w,b)x3 |
//  13:p.w1 14:p.b1 15-18:p.bn1(g,be,mu,va) 19:p.w2 20:p.b2 21-24:p.bn2 |
//  25-36:l (w1,b1,bn1x4,w2,b2,bn2x4) | 37-48:w (same) | 49:main_w
// ---------------------------------------------------------------------------
extern "C" void kernel_launch(void* const* d_in, const int* in_sizes, int n_in,
                              void* d_out, int out_size, void* d_ws, size_t ws_size,
                              hipStream_t stream) {
    (void)in_sizes; (void)n_in; (void)out_size; (void)ws_size;

    const float* X = (const float*)d_in[0];

    // Workspace carve-up (bytes, all 32B-aligned). Total ~107 MB.
    char* base = (char*)d_ws;
    size_t off = 0;
    float*    XPAD = (float*)(base + off);    off += (size_t)XPAD_ELEMS * 4;
    _Float16* T1   = (_Float16*)(base + off); off += (size_t)PIX_ * CN_ * 2;
    _Float16* T2   = (_Float16*)(base + off); off += (size_t)PIX_ * CN_ * 2;
    float*    MB   = (float*)(base + off);    off += (size_t)PIX_ * CN_ * 4;
    float*    BB   = (float*)(base + off);    off += (size_t)PIX_ * CN_ * 4;
    float*    LMAP = (float*)(base + off);    off += (size_t)PIX_ * 4;
    float*    WMAP = (float*)(base + off);    off += (size_t)PIX_ * 4;
    float*    AFF  = (float*)(base + off);    off += 8 * 128 * 4;
    _Float16* WPK  = (_Float16*)(base + off);

    const int FRAG = 4 * KSTEPS_ * 32 * 16;     // 36864 halfs per conv
    _Float16* wp_m0 = WPK + 0 * FRAG, *wp_m1 = WPK + 1 * FRAG, *wp_m2 = WPK + 2 * FRAG;
    _Float16* wp_b0 = WPK + 3 * FRAG, *wp_b1 = WPK + 4 * FRAG, *wp_b2 = WPK + 5 * FRAG;
    _Float16* wp_p1 = WPK + 6 * FRAG, *wp_p2 = WPK + 7 * FRAG, *wp_mn = WPK + 8 * FRAG;

    float* sc[8]; float* sh[8];
    for (int i = 0; i < 8; ++i) { sc[i] = AFF + i * 128; sh[i] = sc[i] + 64; }

    // 1) pad x
    pad_kernel<<<(XPAD_ELEMS + 255) / 256, 256, 0, stream>>>(X, XPAD);

    // 2) pack weights into pi-permuted WMMA B-fragment layout
    const float* wsrc[9] = {
        (const float*)d_in[1], (const float*)d_in[3], (const float*)d_in[5],   // m
        (const float*)d_in[7], (const float*)d_in[9], (const float*)d_in[11],  // b
        (const float*)d_in[13], (const float*)d_in[19],                        // p
        (const float*)d_in[49]                                                 // main
    };
    _Float16* wdst[9] = {wp_m0, wp_m1, wp_m2, wp_b0, wp_b1, wp_b2, wp_p1, wp_p2, wp_mn};
    for (int i = 0; i < 9; ++i)
        pack_w_kernel<<<(FRAG + 255) / 256, 256, 0, stream>>>(wsrc[i], wdst[i]);

    // 3) fold bias / BN into per-channel affine
    prep_affine_kernel<<<1, 64, 0, stream>>>((const float*)d_in[2],  nullptr, nullptr, nullptr, nullptr, sc[0], sh[0]);
    prep_affine_kernel<<<1, 64, 0, stream>>>((const float*)d_in[4],  nullptr, nullptr, nullptr, nullptr, sc[1], sh[1]);
    prep_affine_kernel<<<1, 64, 0, stream>>>((const float*)d_in[6],  nullptr, nullptr, nullptr, nullptr, sc[2], sh[2]);
    prep_affine_kernel<<<1, 64, 0, stream>>>((const float*)d_in[8],  nullptr, nullptr, nullptr, nullptr, sc[3], sh[3]);
    prep_affine_kernel<<<1, 64, 0, stream>>>((const float*)d_in[10], nullptr, nullptr, nullptr, nullptr, sc[4], sh[4]);
    prep_affine_kernel<<<1, 64, 0, stream>>>((const float*)d_in[12], nullptr, nullptr, nullptr, nullptr, sc[5], sh[5]);
    prep_affine_kernel<<<1, 64, 0, stream>>>((const float*)d_in[14],
        (const float*)d_in[15], (const float*)d_in[16], (const float*)d_in[17], (const float*)d_in[18], sc[6], sh[6]);
    prep_affine_kernel<<<1, 64, 0, stream>>>((const float*)d_in[20],
        (const float*)d_in[21], (const float*)d_in[22], (const float*)d_in[23], (const float*)d_in[24], sc[7], sh[7]);

    const int RB = BN_ * HN_;     // 640 blocks, one output row each, 10 waves

    // 4) p branch (convert staging for fp32 x; async staging for f16)
    conv3x3_wmma_kernel<float,    _Float16, STAGE_CONVERT><<<RB, 320, 0, stream>>>(X,  wp_p1, sc[6], sh[6], T1, 100.f, 1);
    conv3x3_wmma_kernel<_Float16, _Float16, STAGE_ASYNC  ><<<RB, 320, 0, stream>>>(T1, wp_p2, sc[7], sh[7], T2, 1.f,   1);

    // 5) heads -> l_map, w_map
    heads_kernel<<<(PIX_ + 255) / 256, 256, 0, stream>>>(T2,
        (const float*)d_in[25], (const float*)d_in[26], (const float*)d_in[27], (const float*)d_in[28],
        (const float*)d_in[29], (const float*)d_in[30], (const float*)d_in[31], (const float*)d_in[32],
        (const float*)d_in[33], (const float*)d_in[34], (const float*)d_in[35], (const float*)d_in[36],
        (const float*)d_in[37], (const float*)d_in[38], (const float*)d_in[39], (const float*)d_in[40],
        (const float*)d_in[41], (const float*)d_in[42], (const float*)d_in[43], (const float*)d_in[44],
        (const float*)d_in[45], (const float*)d_in[46], (const float*)d_in[47], (const float*)d_in[48],
        LMAP, WMAP);

    // 6) m branch: LReLU, LReLU, tanh (TDM staging; last conv emits fp32)
    conv3x3_wmma_kernel<float,    _Float16, STAGE_CONVERT><<<RB, 320, 0, stream>>>(X,  wp_m0, sc[0], sh[0], T1, 1.f, 1);
    conv3x3_wmma_kernel<_Float16, _Float16, STAGE_TDM    ><<<RB, 320, 0, stream>>>(T1, wp_m1, sc[1], sh[1], T2, 1.f, 1);
    conv3x3_wmma_kernel<_Float16, float,    STAGE_TDM    ><<<RB, 320, 0, stream>>>(T2, wp_m2, sc[2], sh[2], MB, 1.f, 2);

    // 7) b branch: LReLU, LReLU, none
    conv3x3_wmma_kernel<float,    _Float16, STAGE_CONVERT><<<RB, 320, 0, stream>>>(X,  wp_b0, sc[3], sh[3], T1, 1.f, 1);
    conv3x3_wmma_kernel<_Float16, _Float16, STAGE_TDM    ><<<RB, 320, 0, stream>>>(T1, wp_b1, sc[4], sh[4], T2, 1.f, 1);
    conv3x3_wmma_kernel<_Float16, float,    STAGE_TDM    ><<<RB, 320, 0, stream>>>(T2, wp_b2, sc[5], sh[5], BB, 1.f, 0);

    // 8) fused bilinear-sample + stride-3 main conv + out*m + bias
    sample_mainconv_kernel<<<MTILES_ / 4, 128, 0, stream>>>(XPAD, LMAP, WMAP, wp_mn, MB, BB,
                                                            (float*)d_out);
}
